// SelfAttention_1434519077194
// MI455X (gfx1250) — matrix operand order repaired and checked by reading, not compiled, not run on hardware
//
#include <hip/hip_runtime.h>
#include <hip/hip_bf16.h>

typedef __bf16 bf16;
typedef __attribute__((ext_vector_type(16))) __bf16 v16bf;
typedef __attribute__((ext_vector_type(8)))  __bf16 v8bf;
typedef __attribute__((ext_vector_type(4)))  __bf16 v4bf;
typedef __attribute__((ext_vector_type(8)))  float  v8f;

union Frag {
    v16bf v;
    v8bf  h[2];
};

__device__ inline v8f v8f_zero() {
    v8f v;
#pragma unroll
    for (int i = 0; i < 8; ++i) v[i] = 0.0f;
    return v;
}

// Build one 16-element bf16 fragment from two 16-byte aligned 8-element halves.
__device__ inline v16bf load_frag(const bf16* p0, const bf16* p1) {
    Frag f;
    f.h[0] = *(const v8bf*)p0;
    f.h[1] = *(const v8bf*)p1;
    return f.v;
}

// Vectorized fp32 -> bf16 (HBM-bound; 16B loads / 8B stores)
__global__ void f32_to_bf16_kernel(const float4* __restrict__ src, bf16* __restrict__ dst, int n4) {
    int i = blockIdx.x * blockDim.x + threadIdx.x;
    if (i < n4) {
        float4 f = src[i];
        v4bf o;
        o[0] = (bf16)f.x; o[1] = (bf16)f.y; o[2] = (bf16)f.z; o[3] = (bf16)f.w;
        *(v4bf*)(dst + (size_t)i * 4) = o;
    }
}

// ---------------------------------------------------------------------------
// Tiled bf16 WMMA GEMM: C[4096 x N] = A[4096 x 1024] * B[1024 x N] + bias
// Block tile 128(M) x 64(N), K staged in 32-slabs through LDS.
// A tile staged with GLOBAL_LOAD_ASYNC_TO_LDS_B128 (ASYNCcnt path, no VGPR
// round-trip); B tile transposed through VGPRs. 8 waves: wave (wm,wn) owns a
// 32x32 sub-tile = 2x2 WMMA 16x16x32 bf16 tiles.
// mode 0: scatter qkv -> q(scaled 1/8)/k/v in [B,H,S,64] bf16
// mode 1: fp32 out + bias
// ---------------------------------------------------------------------------
#define ASTR 40  // padded LDS row stride (elements); 80B rows stay 16B aligned

__global__ void __launch_bounds__(256) wmma_gemm_bf16(
    const bf16* __restrict__ A, const bf16* __restrict__ B,
    const float* __restrict__ bias, int N, int mode,
    float* __restrict__ outF,
    bf16* __restrict__ qh, bf16* __restrict__ kh, bf16* __restrict__ vh)
{
    __shared__ __align__(16) bf16 sA[128 * ASTR];
    __shared__ __align__(16) bf16 sB[64 * 32];   // transposed: [n][k]

    const int t    = threadIdx.x;
    const int wave = t >> 5;
    const int lane = t & 31;
    const int hi   = lane >> 4;   // lane-half select
    const int ml   = lane & 15;
    const int wm   = wave >> 1;   // 0..3
    const int wn   = wave & 1;    // 0..1
    const int mBase = blockIdx.y * 128;
    const int nBase = blockIdx.x * 64;

    v8f acc[2][2];
#pragma unroll
    for (int i = 0; i < 2; ++i)
#pragma unroll
        for (int j = 0; j < 2; ++j) acc[i][j] = v8f_zero();

    const int am = t >> 1;   // A stage: row 0..127
    const int ah = t & 1;    //          16-elem half
    const int bk = t & 31;   // B stage: k 0..31
    const int bg = t >> 5;   //          8-col group

    // LDS byte offset of this thread's A-stage target (addrspace(3) offset
    // lives in the low 32 bits of the generic pointer).
    const unsigned ldsA = (unsigned)(uintptr_t)&sA[am * ASTR + ah * 16];

    for (int kt = 0; kt < 1024; kt += 32) {
        {   // stage A tile 128x32: async global->LDS copy, 32B per thread.
            // INST_OFFSET is added to BOTH the LDS and global address, so one
            // vdst/vaddr pair issues both 16B halves.
            const bf16* gp = A + (size_t)(mBase + am) * 1024 + kt + ah * 16;
            asm volatile(
                "global_load_async_to_lds_b128 %0, %1, off\n\t"
                "global_load_async_to_lds_b128 %0, %1, off offset:16"
                :
                : "v"(ldsA), "v"(gp)
                : "memory");
            if (kt + 32 < 1024) __builtin_prefetch(gp + 32, 0, 1);
        }
        {   // stage B tile 32x64, transposed into [n][k]
            const bf16* gp = B + (size_t)(kt + bk) * N + nBase + bg * 8;
            v8bf b0 = *(const v8bf*)gp;
#pragma unroll
            for (int i = 0; i < 8; ++i) sB[(bg * 8 + i) * 32 + bk] = b0[i];
            if (kt + 32 < 1024) __builtin_prefetch(gp + (size_t)32 * N, 0, 1);
        }
        // __syncthreads only fences DScnt; async LDS writes are tracked by
        // ASYNCcnt and must complete before other waves read the tile.
        asm volatile("s_wait_asynccnt 0x0" ::: "memory");
        __syncthreads();

        // A frag (16x32): lane=M row, elems K = hi*8 + {0..7}, 16 + hi*8 + {0..7}
        v16bf a0 = load_frag(&sA[(wm*32 +  0 + ml) * ASTR + hi*8],
                             &sA[(wm*32 +  0 + ml) * ASTR + 16 + hi*8]);
        v16bf a1 = load_frag(&sA[(wm*32 + 16 + ml) * ASTR + hi*8],
                             &sA[(wm*32 + 16 + ml) * ASTR + 16 + hi*8]);
        // B frag (32x16): lane=N col, elems K = hi*8 + {0..7}, 16 + hi*8 + {0..7}
        v16bf b0 = load_frag(&sB[(wn*32 +  0 + ml) * 32 + hi*8],
                             &sB[(wn*32 +  0 + ml) * 32 + 16 + hi*8]);
        v16bf b1 = load_frag(&sB[(wn*32 + 16 + ml) * 32 + hi*8],
                             &sB[(wn*32 + 16 + ml) * 32 + 16 + hi*8]);

        acc[0][0] = __builtin_amdgcn_wmma_f32_16x16x32_bf16(false, a0, false, b0, (short)0, acc[0][0], false, false);
        acc[0][1] = __builtin_amdgcn_wmma_f32_16x16x32_bf16(false, a0, false, b1, (short)0, acc[0][1], false, false);
        acc[1][0] = __builtin_amdgcn_wmma_f32_16x16x32_bf16(false, a1, false, b0, (short)0, acc[1][0], false, false);
        acc[1][1] = __builtin_amdgcn_wmma_f32_16x16x32_bf16(false, a1, false, b1, (short)0, acc[1][1], false, false);

        __syncthreads();
    }

#pragma unroll
    for (int mt = 0; mt < 2; ++mt) {
#pragma unroll
        for (int nt = 0; nt < 2; ++nt) {
            const int ncol = nBase + wn*32 + nt*16 + ml;
            const float bv = bias[ncol];
#pragma unroll
            for (int r = 0; r < 8; ++r) {
                const int row = mBase + wm*32 + mt*16 + hi*8 + r;
                const float v = acc[mt][nt][r] + bv;
                if (mode == 1) {
                    outF[(size_t)row * N + ncol] = v;
                } else {
                    const int which = ncol >> 10;       // 0=q 1=k 2=v
                    const int d  = ncol & 1023;
                    const int h  = d >> 6;
                    const int dh = d & 63;
                    const int b  = row >> 11;
                    const int s  = row & 2047;
                    const size_t idx = ((size_t)(b * 16 + h) * 2048 + s) * 64 + dh;
                    if (which == 0)      qh[idx] = (bf16)(v * 0.125f);  // fold 1/sqrt(64)
                    else if (which == 1) kh[idx] = (bf16)v;
                    else                 vh[idx] = (bf16)v;
                }
            }
        }
    }
}

// ---------------------------------------------------------------------------
// Flash attention: grid (S/256, B*H), 8 waves; each wave owns 32 query rows
// and streams 32-wide KV tiles with online softmax. Per-wave LDS re-layouts
// the probability tile from C-layout to A-layout for the P@V WMMAs.
// ---------------------------------------------------------------------------
__global__ void __launch_bounds__(256) wmma_flash_attn(
    const bf16* __restrict__ qh, const bf16* __restrict__ kh,
    const bf16* __restrict__ vh, bf16* __restrict__ aout)
{
    __shared__ __align__(16) bf16 sP[8][32 * 32];

    const int bh    = blockIdx.y;           // 0..31 = b*16 + h
    const int lane  = threadIdx.x & 31;
    const int wave  = threadIdx.x >> 5;
    const int hi    = lane >> 4;
    const int ml    = lane & 15;
    const int qbase = blockIdx.x * 256 + wave * 32;

    const bf16* qp = qh + (size_t)bh * 2048 * 64;
    const bf16* kp = kh + (size_t)bh * 2048 * 64;
    const bf16* vp = vh + (size_t)bh * 2048 * 64;
    bf16* sPw = &sP[wave][0];

    // resident Q fragments: 2 row-tiles x 2 K-chunks (d_head=64)
    v16bf qf[2][2];
#pragma unroll
    for (int mt = 0; mt < 2; ++mt)
#pragma unroll
        for (int kc = 0; kc < 2; ++kc) {
            const bf16* p = qp + (size_t)(qbase + mt*16 + ml) * 64 + kc*32 + hi*16;
            qf[mt][kc] = load_frag(p, p + 8);
        }

    float mrow[2][8], lrow[2][8];
    v8f o[2][4];
#pragma unroll
    for (int mt = 0; mt < 2; ++mt) {
#pragma unroll
        for (int r = 0; r < 8; ++r) { mrow[mt][r] = -__builtin_inff(); lrow[mt][r] = 0.0f; }
#pragma unroll
        for (int dt = 0; dt < 4; ++dt) o[mt][dt] = v8f_zero();
    }

    const int numj = qbase / 32 + 1;        // causal: only tiles at/below diagonal
    for (int j = 0; j < numj; ++j) {
        const int kvb = j * 32;

        // K^T fragments (B layout): lane = kv column, elems = head dims
        v16bf kf[2][2];
#pragma unroll
        for (int nt = 0; nt < 2; ++nt)
#pragma unroll
            for (int kc = 0; kc < 2; ++kc) {
                const bf16* p = kp + (size_t)(kvb + nt*16 + ml) * 64 + kc*32 + hi*16;
                kf[nt][kc] = load_frag(p, p + 8);
            }

        // V fragments (B layout): elems = kv rows (strided gather, L2-hot)
        Frag vf[4];
#pragma unroll
        for (int dt = 0; dt < 4; ++dt)
#pragma unroll
            for (int e = 0; e < 16; ++e)
                vf[dt].v[e] = vp[(size_t)(kvb + hi*16 + e) * 64 + dt*16 + ml];

#pragma unroll
        for (int mt = 0; mt < 2; ++mt) {
            v8f s0 = v8f_zero(), s1 = v8f_zero();
            s0 = __builtin_amdgcn_wmma_f32_16x16x32_bf16(false, qf[mt][0], false, kf[0][0], (short)0, s0, false, false);
            s0 = __builtin_amdgcn_wmma_f32_16x16x32_bf16(false, qf[mt][1], false, kf[0][1], (short)0, s0, false, false);
            s1 = __builtin_amdgcn_wmma_f32_16x16x32_bf16(false, qf[mt][0], false, kf[1][0], (short)0, s1, false, false);
            s1 = __builtin_amdgcn_wmma_f32_16x16x32_bf16(false, qf[mt][1], false, kf[1][1], (short)0, s1, false, false);

#pragma unroll
            for (int r = 0; r < 8; ++r) {
                const int row = qbase + mt*16 + hi*8 + r;
                if (kvb + ml      > row) s0[r] = -__builtin_inff();
                if (kvb + 16 + ml > row) s1[r] = -__builtin_inff();

                // row max/sum live across the 16 lanes of this half
                float mx = fmaxf(s0[r], s1[r]);
                mx = fmaxf(mx, __shfl_xor(mx, 1));
                mx = fmaxf(mx, __shfl_xor(mx, 2));
                mx = fmaxf(mx, __shfl_xor(mx, 4));
                mx = fmaxf(mx, __shfl_xor(mx, 8));
                const float mnew = fmaxf(mrow[mt][r], mx);
                const float p0 = __expf(s0[r] - mnew);
                const float p1 = __expf(s1[r] - mnew);
                float rs = p0 + p1;
                rs += __shfl_xor(rs, 1);
                rs += __shfl_xor(rs, 2);
                rs += __shfl_xor(rs, 4);
                rs += __shfl_xor(rs, 8);
                const float sc = __expf(mrow[mt][r] - mnew);
                lrow[mt][r] = lrow[mt][r] * sc + rs;
                mrow[mt][r] = mnew;
#pragma unroll
                for (int dt = 0; dt < 4; ++dt) o[mt][dt][r] *= sc;

                // stash P (C-layout -> row-major 32x32 in per-wave LDS)
                sPw[(mt*16 + hi*8 + r) * 32 + ml]      = (bf16)p0;
                sPw[(mt*16 + hi*8 + r) * 32 + 16 + ml] = (bf16)p1;
            }

            // reload P in A-layout; O += P @ V (DS ops in-order within wave)
            {
                const bf16* p = &sPw[(mt*16 + ml) * 32 + hi*16];
                v16bf pf = load_frag(p, p + 8);
#pragma unroll
                for (int dt = 0; dt < 4; ++dt)
                    o[mt][dt] = __builtin_amdgcn_wmma_f32_16x16x32_bf16(false, pf, false, vf[dt].v, (short)0, o[mt][dt], false, false);
            }
        }
    }

    // normalize and store merged heads: [B*S, 1024] bf16
    const int b = bh >> 4, h = bh & 15;
#pragma unroll
    for (int mt = 0; mt < 2; ++mt)
#pragma unroll
        for (int r = 0; r < 8; ++r) {
            const float inv = 1.0f / lrow[mt][r];
            const int qrow = qbase + mt*16 + hi*8 + r;
#pragma unroll
            for (int dt = 0; dt < 4; ++dt)
                aout[(size_t)(b * 2048 + qrow) * 1024 + h*64 + dt*16 + ml] =
                    (bf16)(o[mt][dt][r] * inv);
        }
}

extern "C" void kernel_launch(void* const* d_in, const int* in_sizes, int n_in,
                              void* d_out, int out_size, void* d_ws, size_t ws_size,
                              hipStream_t stream) {
    (void)in_sizes; (void)n_in; (void)out_size; (void)ws_size;
    const float* x     = (const float*)d_in[0];
    const float* w_in  = (const float*)d_in[1];
    const float* b_in  = (const float*)d_in[2];
    const float* w_out = (const float*)d_in[3];
    const float* b_out = (const float*)d_in[4];
    float* out = (float*)d_out;

    char* ws = (char*)d_ws;
    const size_t MB = 1024 * 1024;
    bf16* xb  = (bf16*)(ws +  0 * MB);  // 8 MB  x as bf16       [4096,1024]
    bf16* wib = (bf16*)(ws +  8 * MB);  // 6 MB  w_in bf16       [1024,3072]
    bf16* wob = (bf16*)(ws + 14 * MB);  // 2 MB  w_out bf16      [1024,1024]
    bf16* qhp = (bf16*)(ws + 16 * MB);  // 8 MB  q (pre-scaled)  [32,2048,64]
    bf16* khp = (bf16*)(ws + 24 * MB);  // 8 MB  k               [32,2048,64]
    bf16* vhp = (bf16*)(ws + 32 * MB);  // 8 MB  v               [32,2048,64]
    bf16* ab  = (bf16*)(ws + 40 * MB);  // 8 MB  attn out        [4096,1024]

    f32_to_bf16_kernel<<<(4096 * 1024 / 4) / 256, 256, 0, stream>>>((const float4*)x,     xb,  4096 * 1024 / 4);
    f32_to_bf16_kernel<<<(1024 * 3072 / 4) / 256, 256, 0, stream>>>((const float4*)w_in,  wib, 1024 * 3072 / 4);
    f32_to_bf16_kernel<<<(1024 * 1024 / 4) / 256, 256, 0, stream>>>((const float4*)w_out, wob, 1024 * 1024 / 4);

    // qkv = x @ w_in + b_in, scattered into per-head q/k/v
    wmma_gemm_bf16<<<dim3(3072 / 64, 4096 / 128), 256, 0, stream>>>(
        xb, wib, b_in, 3072, 0, nullptr, qhp, khp, vhp);

    // causal flash attention
    wmma_flash_attn<<<dim3(2048 / 256, 32), 256, 0, stream>>>(qhp, khp, vhp, ab);

    // out = attn @ w_out + b_out  (fp32 result)
    wmma_gemm_bf16<<<dim3(1024 / 64, 4096 / 128), 256, 0, stream>>>(
        ab, wob, b_out, 1024, 1, out, nullptr, nullptr, nullptr);
}
